// EfficientMultiHeadAttention_19121194402103
// MI455X (gfx1250) — compile-verified, matrix operand order checked by reading, not run on hardware
//
#include <hip/hip_runtime.h>
#include <hip/hip_bf16.h>

// ---------------------------------------------------------------------------
// Windowed MHA for MI455X (gfx1250), bf16 WMMA pipeline.
//   L=8192, B=2, E=1024, H=16, WS=256, D=64
// Pipeline:
//   1) Q/K/V projections: f32 (16384x1024) @ W^T -> bf16, stored (B, L*H, D)
//   2) 1024 windows of 256x64 attention; K/V staged with async global->LDS
//      copies (ASYNCcnt path), WMMA QK^T + softmax + WMMA PV,
//      output scattered into (H,B,L,D) bf16 (reproduces reference reshapes)
//   3) Output projection: bf16 (16384x1024) @ Wo^T + bo -> f32 d_out (L,B,E)
// ---------------------------------------------------------------------------

typedef __bf16 bf16;
typedef __attribute__((ext_vector_type(16))) __bf16 v16bf;
typedef __attribute__((ext_vector_type(8)))  float  v8f;

#define L_SEQ 8192
#define BATCH 2
#define EMB   1024
#define HEADS 16
#define WSZ   256
#define DH    64
#define M_ROWS (L_SEQ * BATCH)          // 16384 GEMM rows

#define LDS_A_STRIDE 40                 // 32 + 8 pad (halves)
#define KB_STRIDE 72                    // 64 + 8 pad
#define VT_STRIDE 264                   // 256 + 8 pad
#define P_STRIDE  264

// Load a 16x32 bf16 fragment (A-layout): lane<16 -> M=lane, K chunks {0..7,16..23};
// lane>=16 -> M=lane-16, K chunks {8..15,24..31}. p0 points at the lane's first
// 8-half chunk (16B aligned); second chunk is +16 halves.
__device__ __forceinline__ v16bf frag_ld(const bf16* p0) {
    union U { uint4 u; bf16 h[8]; };
    U lo, hi;
    lo.u = ((const uint4*)p0)[0];
    hi.u = ((const uint4*)p0)[2];       // +32 bytes = +16 halves
    v16bf r;
#pragma unroll
    for (int i = 0; i < 8; ++i) { r[i] = lo.h[i]; r[8 + i] = hi.h[i]; }
    return r;
}

// Async global->LDS 16-byte copy (gfx1250 GLOBAL_LOAD_ASYNC_TO_LDS_B128,
// tracked by ASYNCcnt). On AMDGPU a generic pointer into LDS carries the LDS
// byte offset in its low 32 bits, which is exactly the VDST operand.
__device__ __forceinline__ void async_cp16(bf16* lds_dst, const bf16* gsrc) {
    unsigned lds_off = (unsigned)(unsigned long long)lds_dst;
    unsigned long long gaddr = (unsigned long long)gsrc;
    asm volatile("global_load_async_to_lds_b128 %0, %1, off"
                 :: "v"(lds_off), "v"(gaddr) : "memory");
}
__device__ __forceinline__ void async_wait0() {
    asm volatile("s_wait_asynccnt 0x0" ::: "memory");
}

// ---------------------------------------------------------------------------
// GEMM: out[m,n] = sum_k A[m,k] * W[n,k] + bias[n]
//   A_F32 : A operand is f32 (converted to bf16 while staging) else bf16
//   MODE  : 0 -> bf16 scatter into (B, L, E) layout (projection stage)
//           1 -> f32 row-major (final output)
// Block: 256 threads (8 waves), tile 128(M) x 128(N), K-step 32.
// Waves 4(M) x 2(N); each wave: 2 M-tiles x 4 N-tiles = 8 WMMA per k-step.
// ---------------------------------------------------------------------------
template <int A_F32, int MODE>
__global__ __launch_bounds__(256)
void gemm_bias_kernel(const void* __restrict__ Aptr,
                      const float* __restrict__ W,
                      const float* __restrict__ bias,
                      void* __restrict__ Out)
{
    __shared__ bf16 As[128 * LDS_A_STRIDE];
    __shared__ bf16 Bs[128 * LDS_A_STRIDE];

    const int tid  = threadIdx.x;
    const int lane = tid & 31;
    const int wid  = tid >> 5;          // 0..7
    const int wm   = wid >> 1;          // 0..3 : 32-row strip
    const int wn   = wid & 1;           // 0..1 : 64-col strip
    const int m0   = blockIdx.x * 128;
    const int n0   = blockIdx.y * 128;

    const float* Af = (const float*)Aptr;
    const bf16*  Ab = (const bf16*)Aptr;

    // staging assignment: thread t stages 16 elements of A row and 16 of B row
    const int sr = tid >> 1;            // 0..127 (row in tile)
    const int sc = (tid & 1) * 16;      // 0/16   (col chunk of 16)

    v8f acc[2][4];
#pragma unroll
    for (int i = 0; i < 2; ++i)
#pragma unroll
        for (int j = 0; j < 4; ++j) acc[i][j] = (v8f){0,0,0,0,0,0,0,0};

    const int lr = lane & 15;
    const int kh = (lane >> 4) << 3;    // 0 or 8

    for (int k0 = 0; k0 < EMB; k0 += 32) {
        // ---- stage A tile (128x32) into LDS as bf16 ----
        if (A_F32) {
            const float* src = Af + (size_t)(m0 + sr) * EMB + k0 + sc;
            float4 f0 = ((const float4*)src)[0];
            float4 f1 = ((const float4*)src)[1];
            float4 f2 = ((const float4*)src)[2];
            float4 f3 = ((const float4*)src)[3];
            bf16* d = As + sr * LDS_A_STRIDE + sc;
            d[0]=(bf16)f0.x; d[1]=(bf16)f0.y; d[2]=(bf16)f0.z; d[3]=(bf16)f0.w;
            d[4]=(bf16)f1.x; d[5]=(bf16)f1.y; d[6]=(bf16)f1.z; d[7]=(bf16)f1.w;
            d[8]=(bf16)f2.x; d[9]=(bf16)f2.y; d[10]=(bf16)f2.z; d[11]=(bf16)f2.w;
            d[12]=(bf16)f3.x; d[13]=(bf16)f3.y; d[14]=(bf16)f3.z; d[15]=(bf16)f3.w;
        } else {
            const bf16* src = Ab + (size_t)(m0 + sr) * EMB + k0 + sc;
            uint4 u0 = ((const uint4*)src)[0];
            uint4 u1 = ((const uint4*)src)[1];
            *(uint4*)(As + sr * LDS_A_STRIDE + sc)     = u0;
            *(uint4*)(As + sr * LDS_A_STRIDE + sc + 8) = u1;
        }
        // ---- stage B tile (128x32) f32 -> bf16 ----
        {
            const float* src = W + (size_t)(n0 + sr) * EMB + k0 + sc;
            float4 f0 = ((const float4*)src)[0];
            float4 f1 = ((const float4*)src)[1];
            float4 f2 = ((const float4*)src)[2];
            float4 f3 = ((const float4*)src)[3];
            bf16* d = Bs + sr * LDS_A_STRIDE + sc;
            d[0]=(bf16)f0.x; d[1]=(bf16)f0.y; d[2]=(bf16)f0.z; d[3]=(bf16)f0.w;
            d[4]=(bf16)f1.x; d[5]=(bf16)f1.y; d[6]=(bf16)f1.z; d[7]=(bf16)f1.w;
            d[8]=(bf16)f2.x; d[9]=(bf16)f2.y; d[10]=(bf16)f2.z; d[11]=(bf16)f2.w;
            d[12]=(bf16)f3.x; d[13]=(bf16)f3.y; d[14]=(bf16)f3.z; d[15]=(bf16)f3.w;
        }
        // prefetch next k-panel (global_prefetch_b8)
        if (k0 + 32 < EMB) {
            if (A_F32)
                __builtin_prefetch(Af + (size_t)(m0 + sr) * EMB + k0 + 32 + sc, 0, 1);
            else
                __builtin_prefetch(Ab + (size_t)(m0 + sr) * EMB + k0 + 32 + sc, 0, 1);
            __builtin_prefetch(W + (size_t)(n0 + sr) * EMB + k0 + 32 + sc, 0, 1);
        }
        __syncthreads();

        // ---- WMMA on 2x4 16x16 tiles per wave, K=32 ----
        v16bf afr[2], bfr[4];
#pragma unroll
        for (int tm = 0; tm < 2; ++tm)
            afr[tm] = frag_ld(As + (wm * 32 + tm * 16 + lr) * LDS_A_STRIDE + kh);
#pragma unroll
        for (int tn = 0; tn < 4; ++tn)
            bfr[tn] = frag_ld(Bs + (wn * 64 + tn * 16 + lr) * LDS_A_STRIDE + kh);
#pragma unroll
        for (int tm = 0; tm < 2; ++tm)
#pragma unroll
            for (int tn = 0; tn < 4; ++tn)
                acc[tm][tn] = __builtin_amdgcn_wmma_f32_16x16x32_bf16(
                    false, afr[tm], false, bfr[tn], (short)0, acc[tm][tn],
                    false, false);
        __syncthreads();
    }

    // ---- epilogue ----
#pragma unroll
    for (int tn = 0; tn < 4; ++tn) {
        const int col = n0 + wn * 64 + tn * 16 + (lane & 15);
        const float bv = bias[col];
#pragma unroll
        for (int tm = 0; tm < 2; ++tm) {
            const int rbase = m0 + wm * 32 + tm * 16 + ((lane >> 4) << 3);
#pragma unroll
            for (int j = 0; j < 8; ++j) {
                const int m = rbase + j;
                const float val = acc[tm][tn][j] + bv;
                if (MODE == 0) {
                    // scatter into (B, L, E) layout so windows are contiguous
                    const int l = m >> 1, b = m & 1;
                    ((bf16*)Out)[(size_t)b * L_SEQ * EMB + (size_t)l * EMB + col] =
                        (bf16)val;
                } else {
                    ((float*)Out)[(size_t)m * EMB + col] = val;
                }
            }
        }
    }
}

// ---------------------------------------------------------------------------
// Windowed attention. One 512-thread block (16 wave32) per 256x64 window.
// K and V staged with async global->LDS copies; V transposed LDS->LDS.
// Wave w owns Q rows [16w, 16w+16): scores in 16 WMMA f32 accumulators,
// softmax via width-16 cross-lane reductions, P relayout via per-wave LDS,
// P@V via WMMA against LDS-resident V^T.
// Output scattered to (H, B, L, D) bf16 layout.
// ---------------------------------------------------------------------------
__global__ __launch_bounds__(512)
void win_attn_kernel(const bf16* __restrict__ Qp, const bf16* __restrict__ Kp,
                     const bf16* __restrict__ Vp, bf16* __restrict__ Oattn)
{
    extern __shared__ bf16 smem[];
    bf16* Kb  = smem;                             // 256 x KB_STRIDE
    bf16* Vt  = Kb + 256 * KB_STRIDE;             // 64  x VT_STRIDE (V^T)
    bf16* Pst = Vt + 64 * VT_STRIDE;              // 16 waves x (16 x P_STRIDE)
    bf16* Vb  = Pst;                              // scratch: row-major V, aliased
                                                  // over P staging (dead by then)

    const int g  = blockIdx.x;                    // 0..1023
    const int b  = g >> 9;                        // batch
    const int hb = g & 511;                       // fused (h, n) window index
    const size_t base = (size_t)b * L_SEQ * EMB + (size_t)hb * WSZ * DH;

    const bf16* Qw = Qp + base;
    const bf16* Kw = Kp + base;
    const bf16* Vw = Vp + base;

    const int tid  = threadIdx.x;
    const int lane = tid & 31;
    const int w    = tid >> 5;                    // wave id 0..15
    const int lr   = lane & 15;
    const int kh   = (lane >> 4) << 3;

    // ---- async-stage K window and (row-major) V window into LDS ----
    {
        const int r  = tid >> 1;                  // 0..255
        const int c0 = (tid & 1) * 32;            // half-row (32 halves = 64B)
#pragma unroll
        for (int i = 0; i < 4; ++i) {
            async_cp16(Kb + r * KB_STRIDE + c0 + 8 * i,
                       Kw + (size_t)r * DH + c0 + 8 * i);
            async_cp16(Vb + r * KB_STRIDE + c0 + 8 * i,
                       Vw + (size_t)r * DH + c0 + 8 * i);
        }
        async_wait0();
    }
    __syncthreads();

    // ---- transpose V (LDS->LDS): Vt[d][f] = Vb[f][d] ----
    {
        const int r  = tid >> 1;
        const int c0 = (tid & 1) * 32;
        union U { uint4 u[4]; bf16 h[32]; } vv;
        const uint4* vs = (const uint4*)(Vb + r * KB_STRIDE + c0);
        vv.u[0] = vs[0]; vv.u[1] = vs[1]; vv.u[2] = vs[2]; vv.u[3] = vs[3];
        __syncthreads();                          // Vb fully read before reuse
#pragma unroll
        for (int i = 0; i < 32; ++i)
            Vt[(c0 + i) * VT_STRIDE + r] = vv.h[i];
    }
    __syncthreads();

    // ---- Q fragments for this wave's 16-row block (direct from global) ----
    const int qr = w * 16;
    v16bf qf[2];
#pragma unroll
    for (int ks = 0; ks < 2; ++ks)
        qf[ks] = frag_ld(Qw + (size_t)(qr + lr) * DH + ks * 32 + kh);

    // ---- S = Q K^T : 16 x 256 strip in 16 accumulators ----
    v8f s[16];
#pragma unroll
    for (int nt = 0; nt < 16; ++nt) {
        v8f c = (v8f){0,0,0,0,0,0,0,0};
#pragma unroll
        for (int ks = 0; ks < 2; ++ks) {
            v16bf kf = frag_ld(Kb + (nt * 16 + lr) * KB_STRIDE + ks * 32 + kh);
            c = __builtin_amdgcn_wmma_f32_16x16x32_bf16(
                    false, qf[ks], false, kf, (short)0, c, false, false);
        }
        s[nt] = c;
    }

    // ---- softmax over 256 columns; rows live across 16-lane groups ----
    const float scale = 0.125f;                   // 1/sqrt(64)
    bf16* Pw = Pst + w * (16 * P_STRIDE);
#pragma unroll
    for (int j = 0; j < 8; ++j) {
        float mx = -3.0e38f;
#pragma unroll
        for (int nt = 0; nt < 16; ++nt) mx = fmaxf(mx, s[nt][j]);
        mx = fmaxf(mx, __shfl_xor(mx, 1, 16));
        mx = fmaxf(mx, __shfl_xor(mx, 2, 16));
        mx = fmaxf(mx, __shfl_xor(mx, 4, 16));
        mx = fmaxf(mx, __shfl_xor(mx, 8, 16));
        float sum = 0.f;
#pragma unroll
        for (int nt = 0; nt < 16; ++nt) {
            float e = __expf((s[nt][j] - mx) * scale);
            s[nt][j] = e;
            sum += e;
        }
        sum += __shfl_xor(sum, 1, 16);
        sum += __shfl_xor(sum, 2, 16);
        sum += __shfl_xor(sum, 4, 16);
        sum += __shfl_xor(sum, 8, 16);
        const float inv = 1.0f / sum;
        const int row = j + ((lane >> 4) << 3);
#pragma unroll
        for (int nt = 0; nt < 16; ++nt)
            Pw[row * P_STRIDE + nt * 16 + (lane & 15)] = (bf16)(s[nt][j] * inv);
    }
    // per-wave LDS RAW on Pw: backend orders DS ops within a wave via dscnt

    // ---- O = P @ V : 16 x 64, K=256 in 8 steps ----
    const int h = hb >> 5, n = hb & 31;
    const size_t obase = (((size_t)h * BATCH + b) * L_SEQ + (size_t)n * WSZ) * DH;
#pragma unroll
    for (int dt = 0; dt < 4; ++dt) {
        v8f c = (v8f){0,0,0,0,0,0,0,0};
#pragma unroll
        for (int ks = 0; ks < 8; ++ks) {
            v16bf pf = frag_ld(Pw + lr * P_STRIDE + ks * 32 + kh);
            v16bf vf = frag_ld(Vt + (dt * 16 + lr) * VT_STRIDE + ks * 32 + kh);
            c = __builtin_amdgcn_wmma_f32_16x16x32_bf16(
                    false, pf, false, vf, (short)0, c, false, false);
        }
        const int d = dt * 16 + (lane & 15);
#pragma unroll
        for (int j = 0; j < 8; ++j) {
            const int row = qr + j + ((lane >> 4) << 3);
            Oattn[obase + (size_t)row * DH + d] = (bf16)c[j];
        }
    }
}

// ---------------------------------------------------------------------------
extern "C" void kernel_launch(void* const* d_in, const int* in_sizes, int n_in,
                              void* d_out, int out_size, void* d_ws, size_t ws_size,
                              hipStream_t stream) {
    const float* q  = (const float*)d_in[0];
    const float* k  = (const float*)d_in[1];
    const float* v  = (const float*)d_in[2];
    const float* Wq = (const float*)d_in[3];  const float* bq = (const float*)d_in[4];
    const float* Wk = (const float*)d_in[5];  const float* bk = (const float*)d_in[6];
    const float* Wv = (const float*)d_in[7];  const float* bv = (const float*)d_in[8];
    const float* Wo = (const float*)d_in[9];  const float* bo = (const float*)d_in[10];
    float* out = (float*)d_out;

    char* wsb = (char*)d_ws;
    const size_t buf = (size_t)BATCH * L_SEQ * EMB * sizeof(bf16);  // 32 MiB
    bf16* Qp = (bf16*)(wsb + 0 * buf);
    bf16* Kp = (bf16*)(wsb + 1 * buf);
    bf16* Vp = (bf16*)(wsb + 2 * buf);
    bf16* Ao = (bf16*)(wsb + 3 * buf);     // (H,B,L,D) attention output

    dim3 gg(M_ROWS / 128, EMB / 128);      // 128 x 8 blocks
    gemm_bias_kernel<1, 0><<<gg, 256, 0, stream>>>(q, Wq, bq, Qp);
    gemm_bias_kernel<1, 0><<<gg, 256, 0, stream>>>(k, Wk, bk, Kp);
    gemm_bias_kernel<1, 0><<<gg, 256, 0, stream>>>(v, Wv, bv, Vp);

    const size_t smem =
        (size_t)(256 * KB_STRIDE + 64 * VT_STRIDE + 16 * 16 * P_STRIDE) * sizeof(bf16);
    win_attn_kernel<<<BATCH * HEADS * (L_SEQ / WSZ), 512, smem, stream>>>(Qp, Kp, Vp, Ao);

    gemm_bias_kernel<0, 1><<<gg, 256, 0, stream>>>(Ao, Wo, bo, out);
}
